// Net_87393994539206
// MI455X (gfx1250) — matrix operand order repaired and checked by reading, not compiled, not run on hardware
//
#include <hip/hip_runtime.h>
#include <hip/hip_bf16.h>
#include <math.h>

// ---------------------------------------------------------------------------
// Problem constants (from reference)
// ---------------------------------------------------------------------------
constexpr int NB   = 65536;  // batch
constexpr int NBLK = 512;    // BLOCK
constexpr int NCU  = 256;    // CU
constexpr int NTAPS = 100;
constexpr float EPS = 1e-5f;

// ---------------------------------------------------------------------------
// WMMA types / helpers (CDNA5 gfx1250, wave32)
// ---------------------------------------------------------------------------
typedef __attribute__((ext_vector_type(16))) __bf16 v16bf;
typedef __attribute__((ext_vector_type(8)))  __bf16 v8bf;
typedef __attribute__((ext_vector_type(8)))  float  v8f;

__device__ inline v8f wmma_bf16(v16bf a, v16bf b, v8f c) {
    // D = A(16x32 bf16) * B(32x16 bf16) + C(16x16 f32)
    return __builtin_amdgcn_wmma_f32_16x16x32_bf16(
        /*neg_a=*/false, a, /*neg_b=*/false, b,
        /*c_mod=*/(short)0, c, /*reuse_a=*/false, /*reuse_b=*/false);
}

// A-fragment (16x32 bf16): lane<16 holds K {k0..k0+15} of row
// M=lane; lane>=16 holds K {k0+16..k0+31} of row M=lane-16.
__device__ inline v16bf frag_a(const __bf16* __restrict__ row, int k0, int sel) {
    const v8bf* p = (const v8bf*)(row + k0 + sel * 16);
    v8bf lo = p[0];       // K chunk +0
    v8bf hi = p[1];       // K chunk +8
    return __builtin_shufflevector(lo, hi, 0, 1, 2, 3, 4, 5, 6, 7,
                                           8, 9, 10, 11, 12, 13, 14, 15);
}

// B-fragment (32x16 bf16): lane<16 holds K {k0..k0+15} of column N=lane,
// lane>=16 holds K {k0+16..k0+31} of column N=lane-16. W is [N,K] row-major
// (we compute x @ W^T), so each lane reads 32 contiguous bytes.
__device__ inline v16bf frag_b(const __bf16* __restrict__ row, int k0, int sel) {
    const v8bf* p = (const v8bf*)(row + k0 + sel * 16);
    v8bf lo = p[0];
    v8bf hi = p[1];
    return __builtin_shufflevector(lo, hi, 0, 1, 2, 3, 4, 5, 6, 7,
                                           8, 9, 10, 11, 12, 13, 14, 15);
}

// ---------------------------------------------------------------------------
// Kernel: FIR 'same' convolution of scaled noise (tiny)
// ---------------------------------------------------------------------------
__global__ __launch_bounds__(256) void fir_kernel(const float* __restrict__ noise,
                                                  const float* __restrict__ fir,
                                                  float* __restrict__ ncp,
                                                  float* __restrict__ out_tail) {
    int i = threadIdx.x;  // 0..255
    const float snr_lin = __powf(10.0f, 0.7f);               // 10^(0.1*7)
    const float noise_std = sqrtf(1.0f / (4.0f * snr_lin));  // rate = 2
    float acc = 0.0f;
    // full conv length 355; 'same' slice starts at (355-256)//2 = 49
#pragma unroll 4
    for (int k = 0; k < NTAPS; ++k) {
        int j = i + 49 - k;
        if (j >= 0 && j < NCU) acc += fir[k] * (noise[j] * noise_std);
    }
    ncp[i] = acc;
    out_tail[i] = acc;  // second tuple output
}

// ---------------------------------------------------------------------------
// Kernel: zero the 4 stats accumulators (contiguous)
// ---------------------------------------------------------------------------
__global__ __launch_bounds__(256) void zero_stats_kernel(float* __restrict__ stats) {
    for (int i = threadIdx.x; i < 4 * NBLK; i += 256) stats[i] = 0.0f;
}

// ---------------------------------------------------------------------------
// Kernel: fp32 -> bf16 bulk convert (8 elements / thread)
// ---------------------------------------------------------------------------
__global__ __launch_bounds__(256) void cvt_bf16_kernel(const float* __restrict__ src,
                                                       __bf16* __restrict__ dst,
                                                       int n8) {
    int i = blockIdx.x * 256 + threadIdx.x;
    if (i >= n8) return;
    const float4* s = (const float4*)src + (size_t)i * 2;
    float4 a = s[0], b = s[1];
    v8bf o;
    o[0] = (__bf16)a.x; o[1] = (__bf16)a.y; o[2] = (__bf16)a.z; o[3] = (__bf16)a.w;
    o[4] = (__bf16)b.x; o[5] = (__bf16)b.y; o[6] = (__bf16)b.z; o[7] = (__bf16)b.w;
    ((v8bf*)dst)[i] = o;
}

// ---------------------------------------------------------------------------
// Kernel: fold BN stats into per-feature scale/shift
// ---------------------------------------------------------------------------
__global__ __launch_bounds__(256) void bn_finalize_kernel(const float* __restrict__ sum,
                                                          const float* __restrict__ sumsq,
                                                          const float* __restrict__ g,
                                                          const float* __restrict__ bt,
                                                          float* __restrict__ scale,
                                                          float* __restrict__ shift) {
    int j = blockIdx.x * 256 + threadIdx.x;
    if (j < NBLK) {
        const float inv_n = 1.0f / (float)NB;
        float mu  = sum[j] * inv_n;
        float var = sumsq[j] * inv_n - mu * mu;
        float rs  = rsqrtf(var + EPS);
        float sc  = g[j] * rs;
        scale[j] = sc;
        shift[j] = bt[j] - mu * sc;
    }
}

// ---------------------------------------------------------------------------
// Kernel: fold BN affine into weights:  bn(h) @ W^T + b
//   Wb[n,k] = bf16(W[n,k] * scale[k]);  bp[n] = b[n] + sum_k shift[k]*W[n,k]
// One block per output row n (coalesced over k).
// ---------------------------------------------------------------------------
__global__ __launch_bounds__(256) void fold_w_kernel(const float* __restrict__ W,
                                                     const float* __restrict__ scale,
                                                     const float* __restrict__ shift,
                                                     const float* __restrict__ b,
                                                     __bf16* __restrict__ Wb,
                                                     float* __restrict__ bp,
                                                     int K) {
    __shared__ float red[256];
    const int n = blockIdx.x;
    const float* wrow = W + (size_t)n * K;
    __bf16* wbrow = Wb + (size_t)n * K;
    float acc = 0.0f;
    for (int k = threadIdx.x; k < K; k += 256) {
        float w = wrow[k];
        wbrow[k] = (__bf16)(w * scale[k]);
        acc += shift[k] * w;
    }
    red[threadIdx.x] = acc;
    __syncthreads();
    for (int s = 128; s > 0; s >>= 1) {
        if (threadIdx.x < s) red[threadIdx.x] += red[threadIdx.x + s];
        __syncthreads();
    }
    if (threadIdx.x == 0) bp[n] = b[n] + red[0];
}

// ---------------------------------------------------------------------------
// Generic bf16 GEMM:  out[M=NB, N=NBLK] = A[M,K] @ W[N,K]^T + bias
//   PRELU_STATS: apply PReLU and accumulate per-column BN sum/sumsq
//   F32OUT:      write fp32 (final layer) instead of bf16
// Block = 256 thr = 8 waves. Each wave owns a 32x64 macro-tile (2 row frags x
// 4 col frags -> 8 WMMAs per K-step on 12 b128 loads, 8 independent
// accumulators to hide WMMA hazards). Block tile 64x256.
// Grid (NB/64, NBLK/256).
// ---------------------------------------------------------------------------
template <int K, bool PRELU_STATS, bool F32OUT>
__global__ __launch_bounds__(256) void gemm_kernel(const __bf16* __restrict__ A,
                                                   const __bf16* __restrict__ W,
                                                   const float* __restrict__ bias,
                                                   const float* __restrict__ alphap,
                                                   __bf16* __restrict__ outb,
                                                   float* __restrict__ outf,
                                                   float* __restrict__ sum,
                                                   float* __restrict__ sumsq) {
    const int lane = threadIdx.x & 31;
    const int wave = threadIdx.x >> 5;
    const int sel  = lane >> 4;
    const int l15  = lane & 15;
    const int r0 = blockIdx.x * 64 + (wave >> 2) * 32;   // wave row base
    const int c0 = blockIdx.y * 256 + (wave & 3) * 64;   // wave col base

    const __bf16* ap0 = A + (size_t)(r0 + l15) * K;
    const __bf16* ap1 = A + (size_t)(r0 + 16 + l15) * K;
    const __bf16* bp0 = W + (size_t)(c0 + l15) * K;
    const __bf16* bp1 = W + (size_t)(c0 + 16 + l15) * K;
    const __bf16* bp2 = W + (size_t)(c0 + 32 + l15) * K;
    const __bf16* bp3 = W + (size_t)(c0 + 48 + l15) * K;

    v8f acc[2][4] = {};
#pragma unroll 2
    for (int k0 = 0; k0 < K; k0 += 32) {
        v16bf a0 = frag_a(ap0, k0, sel);
        v16bf a1 = frag_a(ap1, k0, sel);
        v16bf b0 = frag_b(bp0, k0, sel);
        v16bf b1 = frag_b(bp1, k0, sel);
        v16bf b2 = frag_b(bp2, k0, sel);
        v16bf b3 = frag_b(bp3, k0, sel);
        acc[0][0] = wmma_bf16(a0, b0, acc[0][0]);
        acc[0][1] = wmma_bf16(a0, b1, acc[0][1]);
        acc[0][2] = wmma_bf16(a0, b2, acc[0][2]);
        acc[0][3] = wmma_bf16(a0, b3, acc[0][3]);
        acc[1][0] = wmma_bf16(a1, b0, acc[1][0]);
        acc[1][1] = wmma_bf16(a1, b1, acc[1][1]);
        acc[1][2] = wmma_bf16(a1, b2, acc[1][2]);
        acc[1][3] = wmma_bf16(a1, b3, acc[1][3]);
    }

    const float alpha = PRELU_STATS ? alphap[0] : 0.0f;
#pragma unroll
    for (int t = 0; t < 4; ++t) {
        const int col = c0 + t * 16 + l15;
        const float bv = bias[col];
        float s = 0.f, q = 0.f;
#pragma unroll
        for (int i = 0; i < 8; ++i) {
            const int rowLo = r0 + sel * 8 + i;    // rows of acc[0][t]
            const int rowHi = rowLo + 16;          // rows of acc[1][t]
            float vLo = acc[0][t][i] + bv;
            float vHi = acc[1][t][i] + bv;
            if (PRELU_STATS) {
                vLo = vLo > 0.f ? vLo : alpha * vLo;
                vHi = vHi > 0.f ? vHi : alpha * vHi;
                s += vLo + vHi;
                q += vLo * vLo + vHi * vHi;
            }
            if (F32OUT) {
                outf[(size_t)rowLo * NBLK + col] = vLo;
                outf[(size_t)rowHi * NBLK + col] = vHi;
            } else {
                outb[(size_t)rowLo * NBLK + col] = (__bf16)vLo;
                outb[(size_t)rowHi * NBLK + col] = (__bf16)vHi;
            }
        }
        if (PRELU_STATS) {
            atomicAdd(&sum[col], s);
            atomicAdd(&sumsq[col], q);
        }
    }
}

// ---------------------------------------------------------------------------
// GEMM2 + row-normalize + noise:
//   y = h @ W2b^T + b2p  (BN already folded into W2b/b2p)
//   yc = 16 * y / ||y||_row + ncp
// Block covers a full 32-row x 256-col stripe (8 waves x 2x2 tiles), so the
// L2 row norm is an in-block LDS reduction. Grid (NB/32).
// ---------------------------------------------------------------------------
__global__ __launch_bounds__(256) void gemm2_norm_kernel(const __bf16* __restrict__ h,
                                                         const __bf16* __restrict__ W2b,
                                                         const float* __restrict__ b2p,
                                                         const float* __restrict__ ncp,
                                                         __bf16* __restrict__ yc) {
    __shared__ float rowsq[32];
    const int lane = threadIdx.x & 31;
    const int wave = threadIdx.x >> 5;
    const int sel  = lane >> 4;
    const int l15  = lane & 15;
    const int r0 = blockIdx.x * 32;
    const int c0 = wave * 32;

    const __bf16* a0p = h + (size_t)(r0 + l15) * NBLK;
    const __bf16* a1p = h + (size_t)(r0 + 16 + l15) * NBLK;
    const __bf16* b0p = W2b + (size_t)(c0 + l15) * NBLK;
    const __bf16* b1p = W2b + (size_t)(c0 + 16 + l15) * NBLK;

    v8f c00 = {}, c01 = {}, c10 = {}, c11 = {};
#pragma unroll 2
    for (int k0 = 0; k0 < NBLK; k0 += 32) {
        v16bf a0 = frag_a(a0p, k0, sel);
        v16bf a1 = frag_a(a1p, k0, sel);
        v16bf b0 = frag_b(b0p, k0, sel);
        v16bf b1 = frag_b(b1p, k0, sel);
        c00 = wmma_bf16(a0, b0, c00);
        c01 = wmma_bf16(a0, b1, c01);
        c10 = wmma_bf16(a1, b0, c10);
        c11 = wmma_bf16(a1, b1, c11);
    }

    const int colA = c0 + l15;
    const int colB = c0 + 16 + l15;
    const float biasA = b2p[colA];
    const float biasB = b2p[colB];
#pragma unroll
    for (int i = 0; i < 8; ++i) {
        c00[i] += biasA; c01[i] += biasB;
        c10[i] += biasA; c11[i] += biasB;
    }

    if (threadIdx.x < 32) rowsq[threadIdx.x] = 0.0f;
    __syncthreads();
#pragma unroll
    for (int i = 0; i < 8; ++i) {
        int lr = sel * 8 + i;
        atomicAdd(&rowsq[lr],      c00[i] * c00[i] + c01[i] * c01[i]);  // ds_add_f32
        atomicAdd(&rowsq[lr + 16], c10[i] * c10[i] + c11[i] * c11[i]);
    }
    __syncthreads();

    const float nA = ncp[colA];
    const float nB = ncp[colB];
#pragma unroll
    for (int i = 0; i < 8; ++i) {
        int lr = sel * 8 + i;
        float rnLo = 16.0f * rsqrtf(rowsq[lr]);       // sqrt(CU)=16; / ||y||
        float rnHi = 16.0f * rsqrtf(rowsq[lr + 16]);
        size_t baseLo = (size_t)(r0 + lr) * NCU;
        size_t baseHi = (size_t)(r0 + 16 + lr) * NCU;
        yc[baseLo + colA] = (__bf16)(c00[i] * rnLo + nA);
        yc[baseLo + colB] = (__bf16)(c01[i] * rnLo + nB);
        yc[baseHi + colA] = (__bf16)(c10[i] * rnHi + nA);
        yc[baseHi + colB] = (__bf16)(c11[i] * rnHi + nB);
    }
}

// ---------------------------------------------------------------------------
// Launch
// ---------------------------------------------------------------------------
extern "C" void kernel_launch(void* const* d_in, const int* in_sizes, int n_in,
                              void* d_out, int out_size, void* d_ws, size_t ws_size,
                              hipStream_t stream) {
    (void)in_sizes; (void)n_in; (void)out_size; (void)ws_size;
    const float* x     = (const float*)d_in[0];
    const float* noise = (const float*)d_in[1];
    const float* W1    = (const float*)d_in[2];
    const float* b1    = (const float*)d_in[3];
    const float* a1    = (const float*)d_in[4];
    const float* g1    = (const float*)d_in[5];
    const float* bt1   = (const float*)d_in[6];
    const float* W2    = (const float*)d_in[7];
    const float* b2    = (const float*)d_in[8];
    const float* W3    = (const float*)d_in[9];
    const float* b3    = (const float*)d_in[10];
    const float* a2    = (const float*)d_in[11];
    const float* g2    = (const float*)d_in[12];
    const float* bt2   = (const float*)d_in[13];
    const float* W4    = (const float*)d_in[14];
    const float* b4    = (const float*)d_in[15];
    const float* fir   = (const float*)d_in[16];

    // Workspace layout (bytes):
    //   [0,1KB)          noise_clipped
    //   [1KB,17KB)       stats: sum1,sumsq1,sum2,sumsq2,scale1,shift1,scale2,shift2
    //   [17KB..)         b2p (1KB), b4p (2KB)
    //   [32KB..)         W1b 512KB | W2b 256KB | W3b 256KB | W4b 512KB
    //   [2MB, +64MB)     xb (bf16 NBxNBLK)  -- reused for yc after gemm1
    //   [66MB, +64MB)    h  (bf16 NBxNBLK)  -- reused for d after gemm2
    char* ws = (char*)d_ws;
    float* ncp    = (float*)(ws);
    float* stats  = (float*)(ws + 1024);
    float* sum1   = stats;
    float* sumsq1 = stats + 512;
    float* sum2   = stats + 1024;
    float* sumsq2 = stats + 1536;
    float* scale1 = stats + 2048;
    float* shift1 = stats + 2560;
    float* scale2 = stats + 3072;
    float* shift2 = stats + 3584;
    float* b2p    = (float*)(ws + 17408);
    float* b4p    = (float*)(ws + 18432);
    __bf16* W1b = (__bf16*)(ws + 32768);
    __bf16* W2b = (__bf16*)(ws + 32768 + 524288);
    __bf16* W3b = (__bf16*)(ws + 32768 + 524288 + 262144);
    __bf16* W4b = (__bf16*)(ws + 32768 + 524288 + 262144 + 262144);
    __bf16* xb  = (__bf16*)(ws + (size_t)2 * 1024 * 1024);
    __bf16* h   = (__bf16*)(ws + (size_t)2 * 1024 * 1024 + (size_t)NB * NBLK * 2);
    __bf16* yc  = xb;  // xb dead after gemm1
    __bf16* d   = h;   // h dead after gemm2

    float* out = (float*)d_out;

    // 0) noise path (tiny), zero stats, bf16 conversions
    fir_kernel<<<1, 256, 0, stream>>>(noise, fir, ncp, out + (size_t)NB * NBLK);
    zero_stats_kernel<<<1, 256, 0, stream>>>(stats);
    cvt_bf16_kernel<<<(NB * NBLK / 8 + 255) / 256, 256, 0, stream>>>(x, xb, NB * NBLK / 8);
    cvt_bf16_kernel<<<(NBLK * NBLK / 8 + 255) / 256, 256, 0, stream>>>(W1, W1b, NBLK * NBLK / 8);
    cvt_bf16_kernel<<<(NBLK * NCU / 8 + 255) / 256, 256, 0, stream>>>(W3, W3b, NBLK * NCU / 8);

    // 1) h = prelu(xb @ W1b^T + b1), BN1 stats
    gemm_kernel<NBLK, true, false><<<dim3(NB / 64, NBLK / 256), 256, 0, stream>>>(
        xb, W1b, b1, a1, h, nullptr, sum1, sumsq1);
    bn_finalize_kernel<<<2, 256, 0, stream>>>(sum1, sumsq1, g1, bt1, scale1, shift1);

    // 2) fold BN1 into W2/b2, then yc = 16*(h@W2b^T+b2p)/||.|| + noise
    fold_w_kernel<<<NCU, 256, 0, stream>>>(W2, scale1, shift1, b2, W2b, b2p, NBLK);
    gemm2_norm_kernel<<<dim3(NB / 32), 256, 0, stream>>>(h, W2b, b2p, ncp, yc);

    // 3) d = prelu(yc @ W3b^T + b3), BN2 stats
    gemm_kernel<NCU, true, false><<<dim3(NB / 64, NBLK / 256), 256, 0, stream>>>(
        yc, W3b, b3, a2, d, nullptr, sum2, sumsq2);
    bn_finalize_kernel<<<2, 256, 0, stream>>>(sum2, sumsq2, g2, bt2, scale2, shift2);

    // 4) fold BN2 into W4/b4, then out = d @ W4b^T + b4p (fp32)
    fold_w_kernel<<<NBLK, 256, 0, stream>>>(W4, scale2, shift2, b4, W4b, b4p, NBLK);
    gemm_kernel<NBLK, false, true><<<dim3(NB / 64, NBLK / 256), 256, 0, stream>>>(
        d, W4b, b4p, nullptr, nullptr, out, nullptr, nullptr);
}
